// SSM_22393959481419
// MI455X (gfx1250) — compile-verified, matrix operand order checked by reading, not run on hardware
//
#include <hip/hip_runtime.h>
#include <hip/hip_bf16.h>

typedef __attribute__((ext_vector_type(16))) _Float16 v16h;
typedef __attribute__((ext_vector_type(8)))  float    v8f;

#define N_TOK   16384
#define DM      256
#define DI      512
#define DS      64
#define NH      8
#define HD      64
#define CDIM    640      /* DI + 2*DS */
#define DP      1160     /* DI + CDIM + NH */
#define DP_PAD  1168     /* 73*16 */
#define LC      64       /* chunk length */
#define NC      256      /* chunks = N_TOK/LC */
#define EPSV    1e-5f

/* ------------------------------------------------------------------ */
/* WMMA fragment helpers (CDNA5 16x16x32 f16, wave32; ISA 7.12.2)      */
/* A: lane m=l&15, half=l>>4; VGPR v elems {k0,k0+1},                  */
/*    k0 = 16*(v>>2) + 8*half + 2*(v&3).  B symmetric with n=l&15.     */
/* C/D: col n=l&15, row m = r + 8*half for VGPR r.                     */
/* Per lane the 16 halfs form two contiguous 8-half runs in k, so      */
/* row-major A and transposed-storage B both load as 2 x b128.         */
/* ------------------------------------------------------------------ */
__device__ inline v16h frag_a_rm(const _Float16* M, int ld, int m0, int kb, int lane) {
  int m = m0 + (lane & 15), half = (lane >> 4) & 1;
  v16h f;
#pragma unroll
  for (int v = 0; v < 8; ++v) {
    int k0 = kb + 16 * (v >> 2) + 8 * half + 2 * (v & 3);
    f[2 * v]     = M[(size_t)m * ld + k0];
    f[2 * v + 1] = M[(size_t)m * ld + k0 + 1];
  }
  return f;
}
/* B operand where memory holds the transpose: element (k,n) = M[n*ld+k] */
__device__ inline v16h frag_b_tr(const _Float16* M, int ld, int kb, int n0, int lane) {
  int n = n0 + (lane & 15), half = (lane >> 4) & 1;
  v16h f;
#pragma unroll
  for (int v = 0; v < 8; ++v) {
    int k0 = kb + 16 * (v >> 2) + 8 * half + 2 * (v & 3);
    f[2 * v]     = M[(size_t)n * ld + k0];
    f[2 * v + 1] = M[(size_t)n * ld + k0 + 1];
  }
  return f;
}
__device__ inline v8f wmma16(v16h a, v16h b, v8f c) {
  return __builtin_amdgcn_wmma_f32_16x16x32_f16(false, a, false, b, (short)0, c, false, false);
}
#define V8F_ZERO {0.f,0.f,0.f,0.f,0.f,0.f,0.f,0.f}

/* ------------------------------------------------------------------ */
/* WMMA GEMM: C[M][N] = A[M][K](f16,rm) x W, with W given transposed   */
/* as WT[Npad][K] (rows >= N zero-filled, Npad = 16-aligned ntiles).   */
/* M % 16 == 0, K % 32 == 0. 8 waves/block, one 16x16 tile per wave.   */
/* ------------------------------------------------------------------ */
__global__ __launch_bounds__(256) void wmma_gemm(const _Float16* __restrict__ A,
                                                 const _Float16* __restrict__ BT,
                                                 float* __restrict__ C,
                                                 int M, int N, int K) {
  int lane = threadIdx.x & 31;
  int ntiles = (N + 15) >> 4;
  int wid = blockIdx.x * 8 + (threadIdx.x >> 5);
  int tm = wid / ntiles, tn = wid % ntiles;
  int m0 = tm * 16, n0 = tn * 16;
  if (m0 >= M) return;
  v8f acc = V8F_ZERO;
  for (int kb = 0; kb < K; kb += 32) {
    if (kb + 32 < K) {
      __builtin_prefetch(&A[(size_t)(m0 + (lane & 15)) * K + kb + 32], 0, 1);
      __builtin_prefetch(&BT[(size_t)(n0 + (lane & 15)) * K + kb + 32], 0, 1);
    }
    v16h a = frag_a_rm(A, K, m0, kb, lane);
    v16h b = frag_b_tr(BT, K, kb, n0, lane);
    acc = wmma16(a, b, acc);
  }
  int n = n0 + (lane & 15), half = (lane >> 4) & 1;
  if (n < N) {
#pragma unroll
    for (int r = 0; r < 8; ++r)
      C[(size_t)(m0 + r + 8 * half) * N + n] = acc[r];
  }
}

/* W[K][N] (f32) -> WT[Npad][K] (f16), rows >= N zeroed */
__global__ __launch_bounds__(256) void transpose_pad_f16(const float* __restrict__ W,
                                                         _Float16* __restrict__ WT,
                                                         int K, int N, int Npad) {
  int idx = blockIdx.x * 256 + threadIdx.x;
  if (idx >= Npad * K) return;
  int n = idx / K, k = idx - n * K;
  WT[idx] = (n < N) ? (_Float16)W[(size_t)k * N + n] : (_Float16)0.f;
}

/* ------------------------------------------------------------------ */
/* Serialization: min/max, Hilbert codes, bitonic sort, order/inverse  */
/* ------------------------------------------------------------------ */
__global__ __launch_bounds__(256) void minmax_pos(const float* __restrict__ pos, float* __restrict__ mm) {
  __shared__ float smin[256 * 3], smax[256 * 3];
  int tid = threadIdx.x;
  float mn[3] = {1e30f, 1e30f, 1e30f}, mx[3] = {-1e30f, -1e30f, -1e30f};
  for (int i = tid; i < N_TOK; i += 256)
    for (int d = 0; d < 3; ++d) {
      float v = pos[i * 3 + d];
      mn[d] = fminf(mn[d], v); mx[d] = fmaxf(mx[d], v);
    }
  for (int d = 0; d < 3; ++d) { smin[tid * 3 + d] = mn[d]; smax[tid * 3 + d] = mx[d]; }
  __syncthreads();
  for (int s = 128; s > 0; s >>= 1) {
    if (tid < s)
      for (int d = 0; d < 3; ++d) {
        smin[tid * 3 + d] = fminf(smin[tid * 3 + d], smin[(tid + s) * 3 + d]);
        smax[tid * 3 + d] = fmaxf(smax[tid * 3 + d], smax[(tid + s) * 3 + d]);
      }
    __syncthreads();
  }
  if (tid == 0)
    for (int d = 0; d < 3; ++d) { mm[d] = smin[d]; mm[3 + d] = smax[d]; }
}

__device__ inline unsigned hilbert_code(unsigned g0, unsigned g1, unsigned g2) {
  unsigned x[3] = {g0, g1, g2};
  for (unsigned Q = 512u; Q > 1u; Q >>= 1) {
    unsigned P = Q - 1;
    for (int i = 0; i < 3; ++i) {
      bool cond = (x[i] & Q) != 0;
      unsigned t = (x[0] ^ x[i]) & P;
      unsigned x0n = cond ? (x[0] ^ P) : (x[0] ^ t);
      unsigned xin = cond ? x[i] : (x[i] ^ t);
      x[0] = x0n;
      if (i > 0) x[i] = xin;
    }
  }
  x[1] ^= x[0]; x[2] ^= x[1];
  unsigned t = 0;
  for (unsigned Q = 512u; Q > 1u; Q >>= 1)
    if (x[2] & Q) t ^= (Q - 1);
  x[0] ^= t; x[1] ^= t; x[2] ^= t;
  unsigned code = 0;
  for (int b = 9; b >= 0; --b)
    for (int i = 0; i < 3; ++i)
      code = (code << 1) | ((x[i] >> b) & 1u);
  return code;
}

__global__ __launch_bounds__(256) void hilbert_keys(const float* __restrict__ pos,
                                                    const float* __restrict__ mm,
                                                    unsigned long long* __restrict__ keys) {
  int i = blockIdx.x * 256 + threadIdx.x;
  if (i >= N_TOK) return;
  unsigned g[3];
  for (int d = 0; d < 3; ++d) {
    float gf = (pos[i * 3 + d] - mm[d]) / (mm[3 + d] - mm[d] + 1e-6f) * 1023.f;
    gf = fminf(fmaxf(gf, 0.f), 1023.f);
    g[d] = (unsigned)(int)gf;
  }
  unsigned c0 = hilbert_code(g[0], g[1], g[2]);
  unsigned c1 = hilbert_code(g[1], g[0], g[2]);
  keys[i]         = ((unsigned long long)c0 << 32) | (unsigned)i;
  keys[N_TOK + i] = ((unsigned long long)c1 << 32) | (unsigned)i;
}

__global__ __launch_bounds__(256) void bitonic_step(unsigned long long* __restrict__ keys, int j, int k) {
  int i = blockIdx.x * 256 + threadIdx.x;
  unsigned long long* K = keys + (size_t)blockIdx.y * N_TOK;
  int ixj = i ^ j;
  if (ixj > i) {
    unsigned long long a = K[i], b = K[ixj];
    bool asc = (i & k) == 0;
    if (asc ? (a > b) : (a < b)) { K[i] = b; K[ixj] = a; }
  }
}

__global__ __launch_bounds__(256) void order_inv(const unsigned long long* __restrict__ keys,
                                                 int* __restrict__ orders, int* __restrict__ invs) {
  int i = blockIdx.x * 256 + threadIdx.x;
  if (i >= 2 * N_TOK) return;
  int idx = (int)(unsigned)keys[i];
  orders[i] = idx;
  int b = i >> 14, p = i & (N_TOK - 1);
  invs[(b << 14) + idx] = p;
}

/* ------------------------------------------------------------------ */
/* Per-layer elementwise / normalization kernels                       */
/* ------------------------------------------------------------------ */
/* LN(q)[orders[row]] -> stacked f16; one block per output row, 256 thr */
__global__ __launch_bounds__(256) void ln_gather(const float* __restrict__ q, const int* __restrict__ orders,
                                                 const float* __restrict__ w, const float* __restrict__ b,
                                                 _Float16* __restrict__ out) {
  __shared__ float red[256];
  int row = blockIdx.x, tid = threadIdx.x;
  int src = orders[row];
  float v = q[(size_t)src * DM + tid];
  red[tid] = v; __syncthreads();
  for (int s = 128; s > 0; s >>= 1) { if (tid < s) red[tid] += red[tid + s]; __syncthreads(); }
  float mean = red[0] / DM; __syncthreads();
  float xc = v - mean;
  red[tid] = xc * xc; __syncthreads();
  for (int s = 128; s > 0; s >>= 1) { if (tid < s) red[tid] += red[tid + s]; __syncthreads(); }
  float inv = rsqrtf(red[0] / DM + EPSV);
  out[(size_t)row * DM + tid] = (_Float16)(xc * inv * w[tid] + b[tid]);
}

/* depthwise causal conv(4) + bias + SiLU on cols [512,1152) of zxbcdt */
__global__ __launch_bounds__(256) void conv_silu(const float* __restrict__ zx, const float* __restrict__ cw,
                                                 const float* __restrict__ cb, float* __restrict__ xBC) {
  size_t idx = (size_t)blockIdx.x * 256 + threadIdx.x;
  if (idx >= (size_t)2 * N_TOK * CDIM) return;
  int c = (int)(idx % CDIM);
  size_t bt = idx / CDIM;
  int t = (int)(bt % N_TOK), b = (int)(bt / N_TOK);
  float acc = cb[c];
#pragma unroll
  for (int k = 0; k < 4; ++k) {
    int tt = t - 3 + k;
    if (tt >= 0) acc += cw[c * 4 + k] * zx[((size_t)b * N_TOK + tt) * DP + DI + c];
  }
  xBC[idx] = acc / (1.f + __expf(-acc));
}

/* dt = softplus(raw + bias); also dtA = dt * (-exp(A_log)); [bh][t]    */
__global__ __launch_bounds__(256) void dt_kernel(const float* __restrict__ zx, const float* __restrict__ dtb,
                                                 const float* __restrict__ Alog,
                                                 float* __restrict__ dt_bh, float* __restrict__ dtA) {
  int idx = blockIdx.x * 256 + threadIdx.x;
  if (idx >= 2 * N_TOK * NH) return;
  int h = idx & 7;
  int bt = idx >> 3;
  int t = bt & (N_TOK - 1), b = bt >> 14;
  float x = zx[((size_t)b * N_TOK + t) * DP + (DI + CDIM) + h] + dtb[h];
  float dt = (x > 20.f) ? x : log1pf(__expf(x));
  float A = -__expf(Alog[h]);
  size_t off = ((size_t)(b * NH + h)) * N_TOK + t;
  dt_bh[off] = dt;
  dtA[off] = dt * A;
}

/* in-place inclusive cumsum of dtA within each 64-chunk -> cumL        */
__global__ __launch_bounds__(256) void cumscan(float* __restrict__ L) {
  int g = blockIdx.x * 256 + threadIdx.x;
  if (g >= 2 * NH * NC) return;
  size_t base = (size_t)(g >> 8) * N_TOK + (size_t)(g & 255) * LC;
  float run = 0.f;
  for (int i = 0; i < LC; ++i) { run += L[base + i]; L[base + i] = run; }
}

/* ------------------------------------------------------------------ */
/* Chunked SSD phase 1 (parallel over b,h,chunk): intra-chunk WMMA     */
/* G = C.B^T ; Y_intra = (mask.decay ∘ G) @ (dt*X) ; S_c = Xf^T @ B    */
/* All B operands held transposed in LDS -> contiguous b128 DS loads.  */
/* ------------------------------------------------------------------ */
__global__ __launch_bounds__(256) void chunk_intra(const float* __restrict__ xBC,
                                                   const float* __restrict__ dt_bh,
                                                   const float* __restrict__ cumL,
                                                   float* __restrict__ ybuf,
                                                   float* __restrict__ Sbuf,
                                                   float* __restrict__ Ebuf) {
  int blk = blockIdx.x;           /* bh*NC + c */
  int c = blk & 255, bh = blk >> 8;
  int h = bh & 7, b = bh >> 3;
  __shared__ _Float16 sB [LC * DS];   /* B[t][s]                     */
  __shared__ _Float16 sBT[DS * LC];   /* B[t][s] transposed: [s][t]  */
  __shared__ _Float16 sC [LC * DS];   /* C[t][s]                     */
  __shared__ _Float16 sG [LC * LC];   /* masked/decayed G[t][j]      */
  __shared__ _Float16 sXdT[HD * LC];  /* (dt_j x_j)[p][j]            */
  __shared__ _Float16 sXfT[HD * LC];  /* (f_j dt_j x_j)[p][j]        */
  __shared__ float sL[LC], sdt[LC];
  int tid = threadIdx.x;
  const float* base = xBC + ((size_t)b * N_TOK + (size_t)c * LC) * CDIM;
  for (int e = tid; e < LC * DS; e += 256) {
    int t = e >> 6, s = e & 63;
    const float* row = base + (size_t)t * CDIM;
    float bv = row[DI + s];
    sB[e] = (_Float16)bv;
    sBT[s * LC + t] = (_Float16)bv;
    sC[e] = (_Float16)row[DI + DS + s];
  }
  if (tid < LC) {
    size_t off = (size_t)bh * N_TOK + (size_t)c * LC + tid;
    sdt[tid] = dt_bh[off];
    sL[tid] = cumL[off];
  }
  __syncthreads();
  float Lend = sL[LC - 1];
  for (int e = tid; e < LC * HD; e += 256) {
    int t = e >> 6, p = e & 63;
    float dv = sdt[t] * base[(size_t)t * CDIM + h * HD + p];
    sXdT[p * LC + t] = (_Float16)dv;
    sXfT[p * LC + t] = (_Float16)(__expf(Lend - sL[t]) * dv);
  }
  __syncthreads();
  int lane = tid & 31, wave = tid >> 5;
  /* matmul 1: G[t][j] = sum_s C[t][s] B[j][s]; then causal decay mask */
  for (int sub = 0; sub < 2; ++sub) {
    int tile = wave * 2 + sub;
    int tm = (tile >> 2) * 16, tn = (tile & 3) * 16;
    v8f acc = V8F_ZERO;
    for (int kb = 0; kb < DS; kb += 32) {
      v16h a = frag_a_rm(sC, DS, tm, kb, lane);
      v16h bb = frag_b_tr(sB, DS, kb, tn, lane);   /* (k=s,n=j) = sB[j][s] */
      acc = wmma16(a, bb, acc);
    }
    int j = tn + (lane & 15), half = (lane >> 4) & 1;
#pragma unroll
    for (int r = 0; r < 8; ++r) {
      int t = tm + r + 8 * half;
      float w = (j <= t) ? __expf(sL[t] - sL[j]) : 0.f;
      sG[t * LC + j] = (_Float16)(acc[r] * w);
    }
  }
  __syncthreads();
  /* matmul 2: Y_intra[t][p] = sum_j G[t][j] Xd[j][p]  -> ybuf */
  float* yb = ybuf + ((size_t)b * N_TOK + (size_t)c * LC) * DI + h * HD;
  for (int sub = 0; sub < 2; ++sub) {
    int tile = wave * 2 + sub;
    int tm = (tile >> 2) * 16, tn = (tile & 3) * 16;
    v8f acc = V8F_ZERO;
    for (int kb = 0; kb < LC; kb += 32) {
      v16h a = frag_a_rm(sG, LC, tm, kb, lane);
      v16h bb = frag_b_tr(sXdT, LC, kb, tn, lane); /* (k=j,n=p) = sXdT[p][j] */
      acc = wmma16(a, bb, acc);
    }
    int p = tn + (lane & 15), half = (lane >> 4) & 1;
#pragma unroll
    for (int r = 0; r < 8; ++r)
      yb[(size_t)(tm + r + 8 * half) * DI + p] = acc[r];
  }
  /* matmul 3: S_c[p][s] = sum_j XfT[p][j] B[j][s] -> Sbuf */
  float* Sb = Sbuf + (size_t)blk * (HD * DS);
  for (int sub = 0; sub < 2; ++sub) {
    int tile = wave * 2 + sub;
    int tm = (tile >> 2) * 16, tn = (tile & 3) * 16;
    v8f acc = V8F_ZERO;
    for (int kb = 0; kb < LC; kb += 32) {
      v16h a = frag_a_rm(sXfT, LC, tm, kb, lane);
      v16h bb = frag_b_tr(sBT, LC, kb, tn, lane);  /* (k=j,n=s) = sBT[s][j] */
      acc = wmma16(a, bb, acc);
    }
    int s = tn + (lane & 15), half = (lane >> 4) & 1;
#pragma unroll
    for (int r = 0; r < 8; ++r)
      Sb[(tm + r + 8 * half) * DS + s] = acc[r];
  }
  if (tid == 0) Ebuf[blk] = __expf(Lend);
}

/* ------------------------------------------------------------------ */
/* Chunked SSD phase 2 (parallel over 16 (b,h); sequential over chunks)*/
/* y += e_t ∘ (C @ h_in^T) ; h = E*h + S_c                             */
/* ------------------------------------------------------------------ */
__global__ __launch_bounds__(256) void chunk_state(const float* __restrict__ xBC,
                                                   const float* __restrict__ cumL,
                                                   const float* __restrict__ Sbuf,
                                                   const float* __restrict__ Ebuf,
                                                   float* __restrict__ ybuf) {
  int bh = blockIdx.x;
  int h = bh & 7, b = bh >> 3;
  __shared__ float sh[HD * DS];       /* h[p][s], f32 accumulator      */
  __shared__ _Float16 shF[HD * DS];   /* f16 snapshot, same layout     */
  __shared__ _Float16 sC[LC * DS];    /* C[t][s]                       */
  __shared__ float se[LC];
  int tid = threadIdx.x, lane = tid & 31, wave = tid >> 5;
  for (int e = tid; e < HD * DS; e += 256) sh[e] = 0.f;
  __syncthreads();
  for (int c = 0; c < NC; ++c) {
    const float* base = xBC + ((size_t)b * N_TOK + (size_t)c * LC) * CDIM;
    for (int e = tid; e < LC * DS; e += 256) {
      int t = e >> 6, s = e & 63;
      sC[e] = (_Float16)base[(size_t)t * CDIM + DI + DS + s];
      shF[e] = (_Float16)sh[e];       /* snapshot h (same index space) */
    }
    if (tid < LC) se[tid] = __expf(cumL[(size_t)bh * N_TOK + (size_t)c * LC + tid]);
    __syncthreads();
    float* yb = ybuf + ((size_t)b * N_TOK + (size_t)c * LC) * DI + h * HD;
    for (int sub = 0; sub < 2; ++sub) {
      int tile = wave * 2 + sub;
      int tm = (tile >> 2) * 16, tn = (tile & 3) * 16;
      v8f acc = V8F_ZERO;
      for (int kb = 0; kb < DS; kb += 32) {
        v16h a = frag_a_rm(sC, DS, tm, kb, lane);
        v16h bb = frag_b_tr(shF, DS, kb, tn, lane); /* (k=s,n=p) = shF[p][s] */
        acc = wmma16(a, bb, acc);
      }
      int p = tn + (lane & 15), half = (lane >> 4) & 1;
#pragma unroll
      for (int r = 0; r < 8; ++r) {
        int t = tm + r + 8 * half;
        yb[(size_t)t * DI + p] += se[t] * acc[r];
      }
    }
    __syncthreads();
    float E = Ebuf[bh * NC + c];
    const float* Sb = Sbuf + ((size_t)bh * NC + c) * (HD * DS);
    for (int e = tid; e < HD * DS; e += 256) sh[e] = E * sh[e] + Sb[e];
    __syncthreads();
  }
}

/* y = (y + D*xs) * silu(gate); RMSNorm * norm_w -> f16                 */
__global__ __launch_bounds__(256) void gated_rmsnorm(const float* __restrict__ ybuf,
                                                     const float* __restrict__ xBC,
                                                     const float* __restrict__ zx,
                                                     const float* __restrict__ Dp,
                                                     const float* __restrict__ nw,
                                                     _Float16* __restrict__ yn) {
  __shared__ float red[256];
  __shared__ float vals[DI];
  int row = blockIdx.x, tid = threadIdx.x;
  const float* yr = ybuf + (size_t)row * DI;
  const float* xr = xBC + (size_t)row * CDIM;
  const float* zr = zx + (size_t)row * DP;
  float ss = 0.f;
  for (int ccol = tid; ccol < DI; ccol += 256) {
    int h = ccol >> 6;
    float v = yr[ccol] + Dp[h] * xr[ccol];
    float g = zr[ccol];
    v *= g / (1.f + __expf(-g));
    vals[ccol] = v;
    ss += v * v;
  }
  red[tid] = ss; __syncthreads();
  for (int s = 128; s > 0; s >>= 1) { if (tid < s) red[tid] += red[tid + s]; __syncthreads(); }
  float scale = rsqrtf(red[0] / DI + EPSV);
  for (int ccol = tid; ccol < DI; ccol += 256)
    yn[(size_t)row * DI + ccol] = (_Float16)(vals[ccol] * scale * nw[ccol]);
}

/* q_new = LN( q_old + 0.5*(y0[inv0] + y1[inv1]) )                      */
__global__ __launch_bounds__(256) void combine_ln(const float* __restrict__ qold,
                                                  const float* __restrict__ yout,
                                                  const int* __restrict__ invs,
                                                  const float* __restrict__ w,
                                                  const float* __restrict__ b,
                                                  float* __restrict__ dst) {
  __shared__ float red[256];
  int i = blockIdx.x, d = threadIdx.x;
  int r0 = invs[i], r1 = invs[N_TOK + i];
  float v = qold[(size_t)i * DM + d] +
            0.5f * (yout[(size_t)r0 * DM + d] + yout[((size_t)N_TOK + r1) * DM + d]);
  red[d] = v; __syncthreads();
  for (int s = 128; s > 0; s >>= 1) { if (d < s) red[d] += red[d + s]; __syncthreads(); }
  float mean = red[0] / DM; __syncthreads();
  float xc = v - mean;
  red[d] = xc * xc; __syncthreads();
  for (int s = 128; s > 0; s >>= 1) { if (d < s) red[d] += red[d + s]; __syncthreads(); }
  float inv = rsqrtf(red[0] / DM + EPSV);
  dst[(size_t)i * DM + d] = xc * inv * w[d] + b[d];
}

/* ------------------------------------------------------------------ */
extern "C" void kernel_launch(void* const* d_in, const int* in_sizes, int n_in,
                              void* d_out, int out_size, void* d_ws, size_t ws_size,
                              hipStream_t stream) {
  const float* query  = (const float*)d_in[0];
  const float* qpos   = (const float*)d_in[1];
  const float* pre_w  = (const float*)d_in[2];
  const float* pre_b  = (const float*)d_in[3];
  const float* fin_w  = (const float*)d_in[4];
  const float* fin_b  = (const float*)d_in[5];

  size_t off = 0;
  auto take = [&](size_t bytes) -> void* {
    void* r = (char*)d_ws + off;
    off += (bytes + 255) & ~(size_t)255;
    return r;
  };
  float*              mm      = (float*)take(6 * 4);
  unsigned long long* keys    = (unsigned long long*)take((size_t)2 * N_TOK * 8);
  int*                orders  = (int*)take((size_t)2 * N_TOK * 4);
  int*                invs    = (int*)take((size_t)2 * N_TOK * 4);
  float*              q_buf   = (float*)take((size_t)N_TOK * DM * 4);
  _Float16*           stacked = (_Float16*)take((size_t)2 * N_TOK * DM * 2);
  _Float16*           w_in16t = (_Float16*)take((size_t)DP_PAD * DM * 2);
  _Float16*           w_out16t= (_Float16*)take((size_t)DM * DI * 2);
  float*              zxbcdt  = (float*)take((size_t)2 * N_TOK * DP * 4);
  float*              xBC     = (float*)take((size_t)2 * N_TOK * CDIM * 4);
  float*              dt_bh   = (float*)take((size_t)2 * NH * N_TOK * 4);
  float*              cumL    = (float*)take((size_t)2 * NH * N_TOK * 4);
  float*              y_buf   = (float*)take((size_t)2 * N_TOK * DI * 4);
  _Float16*           yn16    = (_Float16*)take((size_t)2 * N_TOK * DI * 2);
  float*              Sbuf    = (float*)take((size_t)2 * NH * NC * HD * DS * 4);
  float*              Ebuf    = (float*)take((size_t)2 * NH * NC * 4);
  float*              yout    = (float*)take((size_t)2 * N_TOK * DM * 4);
  if (off > ws_size) return;  /* workspace too small; nothing safe to do */

  /* --- serialization (once) --- */
  minmax_pos<<<1, 256, 0, stream>>>(qpos, mm);
  hilbert_keys<<<N_TOK / 256, 256, 0, stream>>>(qpos, mm, keys);
  for (int k = 2; k <= N_TOK; k <<= 1)
    for (int j = k >> 1; j > 0; j >>= 1)
      bitonic_step<<<dim3(N_TOK / 256, 2), 256, 0, stream>>>(keys, j, k);
  order_inv<<<2 * N_TOK / 256, 256, 0, stream>>>(keys, orders, invs);

  /* --- two Mamba2 layers --- */
  for (int l = 0; l < 2; ++l) {
    const float* in_proj = (const float*)d_in[6 + 8 * l];
    const float* conv_w  = (const float*)d_in[7 + 8 * l];
    const float* conv_b  = (const float*)d_in[8 + 8 * l];
    const float* dtbias  = (const float*)d_in[9 + 8 * l];
    const float* Alog    = (const float*)d_in[10 + 8 * l];
    const float* Dp      = (const float*)d_in[11 + 8 * l];
    const float* norm_w  = (const float*)d_in[12 + 8 * l];
    const float* outproj = (const float*)d_in[13 + 8 * l];
    const float* qsrc = (l == 0) ? query : q_buf;
    float* dst = (l == 1) ? (float*)d_out : q_buf;

    transpose_pad_f16<<<(DP_PAD * DM + 255) / 256, 256, 0, stream>>>(in_proj, w_in16t, DM, DP, DP_PAD);
    transpose_pad_f16<<<(DM * DI + 255) / 256, 256, 0, stream>>>(outproj, w_out16t, DI, DM, DM);

    ln_gather<<<2 * N_TOK, 256, 0, stream>>>(qsrc, orders, pre_w, pre_b, stacked);

    { /* in_proj GEMM: [32768 x 256] @ [256 x 1160] */
      int tiles = (2 * N_TOK / 16) * ((DP + 15) / 16);
      wmma_gemm<<<(tiles + 7) / 8, 256, 0, stream>>>(stacked, w_in16t, zxbcdt, 2 * N_TOK, DP, DM);
    }

    conv_silu<<<(unsigned)(((size_t)2 * N_TOK * CDIM + 255) / 256), 256, 0, stream>>>(zxbcdt, conv_w, conv_b, xBC);
    dt_kernel<<<(2 * N_TOK * NH + 255) / 256, 256, 0, stream>>>(zxbcdt, dtbias, Alog, dt_bh, cumL);
    cumscan<<<(2 * NH * NC + 255) / 256, 256, 0, stream>>>(cumL);

    chunk_intra<<<2 * NH * NC, 256, 0, stream>>>(xBC, dt_bh, cumL, y_buf, Sbuf, Ebuf);
    chunk_state<<<2 * NH, 256, 0, stream>>>(xBC, cumL, Sbuf, Ebuf, y_buf);

    gated_rmsnorm<<<2 * N_TOK, 256, 0, stream>>>(y_buf, xBC, zxbcdt, Dp, norm_w, yn16);

    { /* out_proj GEMM: [32768 x 512] @ [512 x 256] */
      int tiles = (2 * N_TOK / 16) * (DM / 16);
      wmma_gemm<<<(tiles + 7) / 8, 256, 0, stream>>>(yn16, w_out16t, yout, 2 * N_TOK, DM, DI);
    }

    combine_ln<<<N_TOK, 256, 0, stream>>>(qsrc, yout, invs, fin_w, fin_b, dst);
  }
}
